// EfficientCrossAttention_36447092474494
// MI455X (gfx1250) — matrix operand order repaired and checked by reading, not compiled, not run on hardware
//
#include <hip/hip_runtime.h>
#include <hip/hip_bf16.h>

typedef __attribute__((ext_vector_type(16))) _Float16 v16h;
typedef __attribute__((ext_vector_type(2)))  _Float16 v2h;
typedef __attribute__((ext_vector_type(8)))  float    v8f;
typedef __attribute__((ext_vector_type(4)))  int      v4i;

#define BB 4
#define CC 512
#define LL 2048
#define HH 8
#define DD 64

#if __has_builtin(__builtin_amdgcn_global_load_async_to_lds_b128)
#define HAVE_ASYNC_LDS 1
#endif

// copy 16 bytes global -> LDS (async-to-LDS on gfx1250)
__device__ __forceinline__ void copy16_g2l(const _Float16* g, _Float16* l) {
#ifdef HAVE_ASYNC_LDS
  __builtin_amdgcn_global_load_async_to_lds_b128(
      (__attribute__((address_space(1))) v4i*)(uintptr_t)g,
      (__attribute__((address_space(3))) v4i*)(unsigned int)(uintptr_t)l,
      0, 0);
#else
  *(float4*)l = *(const float4*)g;
#endif
}

__device__ __forceinline__ void wait_async0() {
#ifdef HAVE_ASYNC_LDS
#if __has_builtin(__builtin_amdgcn_s_wait_asynccnt)
  __builtin_amdgcn_s_wait_asynccnt(0);
#else
  asm volatile("s_wait_asynccnt 0x0" ::: "memory");
#endif
#endif
}

__device__ __forceinline__ v8f wmma16(v16h a, v16h b, v8f c) {
  return __builtin_amdgcn_wmma_f32_16x16x32_f16(false, a, false, b, (short)0, c, false, false);
}

// A fragment from LDS tile row-major [rows][ldsStride], rows r0..r0+15, K 0..31 of tile
__device__ __forceinline__ v16h lds_a_frag(const _Float16* base, int ldsStride, int lane) {
  int m  = lane & 15;
  int kb = (lane & 16) ? 8 : 0;
  const _Float16* p = base + m * ldsStride + kb;
  v16h a;
#pragma unroll
  for (int e = 0; e < 8; ++e) a[e] = p[e];
#pragma unroll
  for (int e = 0; e < 8; ++e) a[8 + e] = p[16 + e];
  return a;
}

// B fragment from K-contiguous storage: p points at [col n][8h]; elements k 8h..8h+7 and 16+8h..16+8h+7
__device__ __forceinline__ v16h lds_kcontig_frag(const _Float16* p) {
  v16h b;
#pragma unroll
  for (int e = 0; e < 8; ++e) b[e] = p[e];
#pragma unroll
  for (int e = 0; e < 8; ++e) b[8 + e] = p[16 + e];
  return b;
}

// A fragment straight from global row-major [*, lda] (contiguous 16B runs per lane)
__device__ __forceinline__ v16h glb_a_frag(const _Float16* A, int lda, int m0, int k0, int lane) {
  int m  = m0 + (lane & 15);
  int kb = k0 + ((lane & 16) ? 8 : 0);
  const _Float16* p = A + (size_t)m * lda + kb;
  v16h a;
#pragma unroll
  for (int e = 0; e < 8; ++e) a[e] = p[e];
#pragma unroll
  for (int e = 0; e < 8; ++e) a[8 + e] = p[16 + e];
  return a;
}

// ---------------------------------------------------------------------------
// 1) transpose + pos-encoding + layernorm.  x:(B,C,L) -> out:(B,L,C)
// ---------------------------------------------------------------------------
__global__ __launch_bounds__(256) void prep_norm(const float* __restrict__ x,
                                                 const float* __restrict__ pos,
                                                 const float* __restrict__ gamma,
                                                 const float* __restrict__ beta,
                                                 float* __restrict__ of32,
                                                 _Float16* __restrict__ oh) {
  __shared__ float tile[16][513];
  __shared__ float smu[16], srs[16];
  int b  = blockIdx.x / (LL / 16);
  int t0 = (blockIdx.x % (LL / 16)) * 16;
  const float* xb = x + (size_t)b * CC * LL;

  int tl = threadIdx.x & 15;
  int cg = threadIdx.x >> 4;
  for (int c = cg; c < CC; c += 16)
    tile[tl][c] = xb[(size_t)c * LL + t0 + tl];
  __syncthreads();

  for (int idx = threadIdx.x; idx < 16 * CC; idx += 256) {
    int l = idx >> 9, c = idx & (CC - 1);
    tile[l][c] += pos[(size_t)(t0 + l) * CC + c];
  }
  __syncthreads();

  int wave = threadIdx.x >> 5, lane = threadIdx.x & 31;
  int row = wave * 2 + (lane >> 4);
  int lc  = lane & 15;
  float sum = 0.f, sq = 0.f;
  for (int c = lc; c < CC; c += 16) { float v = tile[row][c]; sum += v; sq += v * v; }
#pragma unroll
  for (int mask = 1; mask <= 8; mask <<= 1) {
    sum += __shfl_xor(sum, mask, 32);
    sq  += __shfl_xor(sq,  mask, 32);
  }
  float mu = sum * (1.f / CC);
  float var = sq * (1.f / CC) - mu * mu;
  float rs = rsqrtf(var + 1e-5f);
  if (lc == 0) { smu[row] = mu; srs[row] = rs; }
  __syncthreads();

  for (int idx = threadIdx.x; idx < 16 * CC; idx += 256) {
    int l = idx >> 9, c = idx & (CC - 1);
    float v = (tile[l][c] - smu[l]) * srs[l] * gamma[c] + beta[c];
    size_t o = ((size_t)b * LL + t0 + l) * CC + c;
    if (of32) of32[o] = v;
    oh[o] = (_Float16)v;
  }
}

// ---------------------------------------------------------------------------
// 2) f32 -> f16 weight conversion
// ---------------------------------------------------------------------------
__global__ __launch_bounds__(256) void f32_to_f16(const float* __restrict__ s,
                                                  _Float16* __restrict__ d, int n) {
  int i = blockIdx.x * 256 + threadIdx.x;
  if (i < n) d[i] = (_Float16)s[i];
}

// ---------------------------------------------------------------------------
// GEMM core: macro tile 128x128, 8 waves (4M x 2N), wave tile 32x64.
// Double-buffered LDS: A staged async as-is, B staged transposed Bt[n][k].
// ---------------------------------------------------------------------------
#define ALD 40   // A lds row stride (halves): 20 DW, conflict-free
#define BLD 36   // Bt lds row stride (halves): 18 DW, conflict-free

template <int EPI>   // 0: f16 out + bias ; 1: f32 out + bias + residual + (B,C,L) transpose
__global__ __launch_bounds__(256) void gemm_wmma(const _Float16* __restrict__ X,
                                                 const _Float16* __restrict__ W,
                                                 const float* __restrict__ bias,
                                                 const float* __restrict__ resid,
                                                 void* __restrict__ Yv,
                                                 int N, int K) {
  __shared__ _Float16 Alds[2][128][ALD];
  __shared__ _Float16 Blds[2][128][BLD];

  const int t = threadIdx.x;
  int wave = t >> 5, lane = t & 31;
  int tM0 = blockIdx.x * 128;
  int tN0 = blockIdx.y * 128;
  int wM = (wave & 3) * 32;        // wave row offset in macro tile
  int wN = (wave >> 2) * 64;       // wave col offset in macro tile

  auto stage = [&](int k0, int buf) {
    // A (128x32): 512 x 16B chunks, 2 per thread (async to LDS)
#pragma unroll
    for (int i = 0; i < 2; ++i) {
      int c = t + i * 256;
      int row = c >> 2, kh = c & 3;
      copy16_g2l(X + (size_t)(tM0 + row) * K + k0 + kh * 8, &Alds[buf][row][kh * 8]);
    }
    // B transposed: Bt[n][k]
    const v2h* Wd = (const v2h*)(W + (size_t)k0 * N + tN0);
    const int ldw = N >> 1;
#pragma unroll
    for (int i = 0; i < 8; ++i) {
      int j = t + i * 256;           // 0..2047 dwords
      int k = j >> 6, ndw = j & 63;
      v2h d = Wd[(size_t)k * ldw + ndw];
      Blds[buf][2 * ndw][k]     = d[0];
      Blds[buf][2 * ndw + 1][k] = d[1];
    }
  };

  v8f acc[2][4];
#pragma unroll
  for (int i = 0; i < 2; ++i)
#pragma unroll
    for (int j = 0; j < 4; ++j) acc[i][j] = {};

  const int nk = K / 32;
  stage(0, 0);
  wait_async0();
  __syncthreads();

  for (int ki = 0; ki < nk; ++ki) {
    int buf = ki & 1;
    if (ki + 1 < nk) stage((ki + 1) * 32, buf ^ 1);

    v16h a0 = lds_a_frag(&Alds[buf][wM][0],      ALD, lane);
    v16h a1 = lds_a_frag(&Alds[buf][wM + 16][0], ALD, lane);
    int kk = (lane & 16) ? 8 : 0;
#pragma unroll
    for (int ni = 0; ni < 4; ++ni) {
      v16h b = lds_kcontig_frag(&Blds[buf][wN + ni * 16 + (lane & 15)][kk]);
      acc[0][ni] = wmma16(a0, b, acc[0][ni]);
      acc[1][ni] = wmma16(a1, b, acc[1][ni]);
    }
    wait_async0();
    __syncthreads();
  }

  int n  = lane & 15;
  int rs = (lane & 16) ? 8 : 0;
#pragma unroll
  for (int mi = 0; mi < 2; ++mi)
#pragma unroll
    for (int ni = 0; ni < 4; ++ni) {
      int col = tN0 + wN + ni * 16 + n;
      float bv = bias[col];
#pragma unroll
      for (int v = 0; v < 8; ++v) {
        int row = tM0 + wM + mi * 16 + rs + v;
        float r = acc[mi][ni][v] + bv;
        if (EPI == 0) {
          ((_Float16*)Yv)[(size_t)row * N + col] = (_Float16)r;
        } else {
          r += resid[(size_t)row * CC + col];
          int bb = row >> 11;            // / LL
          int l  = row & (LL - 1);
          ((float*)Yv)[((size_t)bb * CC + col) * LL + l] = r;
        }
      }
    }
}

// ---------------------------------------------------------------------------
// 4) flash attention.  Q,K,V: (B,L,C) f16; out AO: (B,L,C) f16
//    block = (b, h, 128 query rows); wave = 16 query rows; double-buffered K/V
// ---------------------------------------------------------------------------
#define KLD 72   // Klds row stride (halves): 36 DW
#define VLD 36   // Vt  row stride (halves): 18 DW

__global__ __launch_bounds__(256) void attn(const _Float16* __restrict__ Q,
                                            const _Float16* __restrict__ K,
                                            const _Float16* __restrict__ V,
                                            _Float16* __restrict__ AO) {
  __shared__ _Float16 Klds[2][32][KLD];   // [key][d]
  __shared__ _Float16 Vt[2][64][VLD];     // [d][key]
  __shared__ _Float16 plds[8][16][40];    // per-wave P transpose buffer

  const int QT = LL / 128;
  const int t = threadIdx.x;
  int wave = t >> 5, lane = t & 31;
  int qt = blockIdx.x % QT;
  int bh = blockIdx.x / QT;
  int h = bh % HH, b = bh / HH;
  int q0 = qt * 128 + wave * 16;

  const _Float16* Qb = Q + (size_t)b * LL * CC + h * DD;
  const _Float16* Kb = K + (size_t)b * LL * CC + h * DD;
  const _Float16* Vb = V + (size_t)b * LL * CC + h * DD;

  auto stage_kv = [&](int kt, int buf) {
    // K tile (32 keys x 64 d), row-major: 256 x 16B chunks, 1 per thread (async)
    int row = t >> 3, ch = t & 7;
    copy16_g2l(Kb + (size_t)(kt + row) * CC + ch * 8, &Klds[buf][row][ch * 8]);
    // V tile transposed: Vt[d][key]
    const v2h* Vp = (const v2h*)(Vb + (size_t)kt * CC);
    const int ldv = CC >> 1;
#pragma unroll
    for (int i = 0; i < 4; ++i) {
      int j = t + i * 256;             // 0..1023 dwords
      int key = j >> 5, ndw = j & 31;
      v2h d = Vp[(size_t)key * ldv + ndw];
      Vt[buf][2 * ndw][key]     = d[0];
      Vt[buf][2 * ndw + 1][key] = d[1];
    }
  };

  v16h qa0 = glb_a_frag(Qb, CC, q0, 0, lane);
  v16h qa1 = glb_a_frag(Qb, CC, q0, 32, lane);
#pragma unroll
  for (int e = 0; e < 16; ++e) { qa0[e] = qa0[e] * (_Float16)0.125f; qa1[e] = qa1[e] * (_Float16)0.125f; }

  v8f acc[4];
#pragma unroll
  for (int d = 0; d < 4; ++d) acc[d] = {};
  float mrow[8], lrow[8];
#pragma unroll
  for (int v = 0; v < 8; ++v) { mrow[v] = -1e30f; lrow[v] = 0.f; }

  int n  = lane & 15;
  int rs = (lane & 16) ? 8 : 0;
  int hk = (lane & 16) ? 8 : 0;

  const int nkt = LL / 32;
  stage_kv(0, 0);
  wait_async0();
  __syncthreads();

  for (int ki = 0; ki < nkt; ++ki) {
    int buf = ki & 1;
    if (ki + 1 < nkt) stage_kv((ki + 1) * 32, buf ^ 1);

    // S = (Q/8) @ K^T   (16 x 32)
    v8f s[2]; s[0] = {}; s[1] = {};
#pragma unroll
    for (int ni = 0; ni < 2; ++ni) {
      const _Float16* kp = &Klds[buf][ni * 16 + n][hk];
      v16h b0 = lds_kcontig_frag(kp);
      v16h b1 = lds_kcontig_frag(kp + 32);
      s[ni] = wmma16(qa0, b0, s[ni]);
      s[ni] = wmma16(qa1, b1, s[ni]);
    }
#pragma unroll
    for (int ni = 0; ni < 2; ++ni)
#pragma unroll
      for (int v = 0; v < 8; ++v)
        s[ni][v] = fminf(50.f, fmaxf(-50.f, s[ni][v]));

    // online softmax (rows live across 16-lane halves)
    float p0[8], p1[8];
#pragma unroll
    for (int v = 0; v < 8; ++v) {
      float tmax = fmaxf(s[0][v], s[1][v]);
#pragma unroll
      for (int mask = 1; mask <= 8; mask <<= 1)
        tmax = fmaxf(tmax, __shfl_xor(tmax, mask, 32));
      float mn = fmaxf(mrow[v], tmax);
      float al = __expf(mrow[v] - mn);
      mrow[v] = mn;
      p0[v] = __expf(s[0][v] - mn);
      p1[v] = __expf(s[1][v] - mn);
      float rsum = p0[v] + p1[v];
#pragma unroll
      for (int mask = 1; mask <= 8; mask <<= 1)
        rsum += __shfl_xor(rsum, mask, 32);
      lrow[v] = lrow[v] * al + rsum;
#pragma unroll
      for (int d = 0; d < 4; ++d) acc[d][v] *= al;
    }

    // P: C-layout -> A-layout through per-wave LDS
#pragma unroll
    for (int v = 0; v < 8; ++v) {
      plds[wave][rs + v][n]      = (_Float16)p0[v];
      plds[wave][rs + v][16 + n] = (_Float16)p1[v];
    }
    __builtin_amdgcn_wave_barrier();
    v16h pa;
    {
      int m  = lane & 15;
      int kb = (lane & 16) ? 8 : 0;
      const _Float16* pp = &plds[wave][m][0];
#pragma unroll
      for (int e = 0; e < 8; ++e) pa[e] = pp[kb + e];
#pragma unroll
      for (int e = 0; e < 8; ++e) pa[8 + e] = pp[kb + 16 + e];
    }
    __builtin_amdgcn_wave_barrier();

    // O += P @ V   (16 x 64) from transposed V tile
#pragma unroll
    for (int di = 0; di < 4; ++di) {
      v16h bv = lds_kcontig_frag(&Vt[buf][di * 16 + n][hk]);
      acc[di] = wmma16(pa, bv, acc[di]);
    }
    wait_async0();
    __syncthreads();
  }

  // normalize + store (rcp instead of 32 divides)
#pragma unroll
  for (int v = 0; v < 8; ++v) lrow[v] = __frcp_rn(lrow[v]);
#pragma unroll
  for (int di = 0; di < 4; ++di)
#pragma unroll
    for (int v = 0; v < 8; ++v) {
      float o = acc[di][v] * lrow[v];
      int row = q0 + rs + v;
      int col = h * DD + di * 16 + n;
      AO[((size_t)b * LL + row) * CC + col] = (_Float16)o;
    }
}

// ---------------------------------------------------------------------------
extern "C" void kernel_launch(void* const* d_in, const int* in_sizes, int n_in,
                              void* d_out, int out_size, void* d_ws, size_t ws_size,
                              hipStream_t stream) {
  (void)in_sizes; (void)n_in; (void)out_size; (void)ws_size;
  const float* query = (const float*)d_in[0];
  const float* keyv  = (const float*)d_in[1];
  const float* pos   = (const float*)d_in[2];
  const float* qg    = (const float*)d_in[3];
  const float* qbeta = (const float*)d_in[4];
  const float* kg    = (const float*)d_in[5];
  const float* kbeta = (const float*)d_in[6];
  const float* Wq    = (const float*)d_in[7];
  const float* bq    = (const float*)d_in[8];
  const float* Wk    = (const float*)d_in[9];
  const float* bk    = (const float*)d_in[10];
  const float* Wv    = (const float*)d_in[11];
  const float* bv    = (const float*)d_in[12];
  const float* Wo    = (const float*)d_in[13];
  const float* bo    = (const float*)d_in[14];

  char* ws = (char*)d_ws;
  const size_t szQN32 = (size_t)BB * LL * CC * 4;
  const size_t szH    = (size_t)BB * LL * CC * 2;
  const size_t szW    = (size_t)CC * CC * 2;
  float*    qn32 = (float*)ws;    ws += szQN32;
  _Float16* qnh  = (_Float16*)ws; ws += szH;
  _Float16* kvnh = (_Float16*)ws; ws += szH;
  _Float16* Wqh  = (_Float16*)ws; ws += szW;
  _Float16* Wkh  = (_Float16*)ws; ws += szW;
  _Float16* Wvh  = (_Float16*)ws; ws += szW;
  _Float16* Woh  = (_Float16*)ws; ws += szW;
  _Float16* Qh   = (_Float16*)ws; ws += szH;
  _Float16* Kh   = (_Float16*)ws; ws += szH;
  _Float16* Vh   = (_Float16*)ws; ws += szH;
  _Float16* AOh  = (_Float16*)ws; ws += szH;

  prep_norm<<<BB * (LL / 16), 256, 0, stream>>>(query, pos, qg, qbeta, qn32, qnh);
  prep_norm<<<BB * (LL / 16), 256, 0, stream>>>(keyv,  pos, kg, kbeta, nullptr, kvnh);

  const int wn = CC * CC;
  f32_to_f16<<<(wn + 255) / 256, 256, 0, stream>>>(Wq, Wqh, wn);
  f32_to_f16<<<(wn + 255) / 256, 256, 0, stream>>>(Wk, Wkh, wn);
  f32_to_f16<<<(wn + 255) / 256, 256, 0, stream>>>(Wv, Wvh, wn);
  f32_to_f16<<<(wn + 255) / 256, 256, 0, stream>>>(Wo, Woh, wn);

  dim3 gg(BB * LL / 128, CC / 128);
  gemm_wmma<0><<<gg, 256, 0, stream>>>(qnh,  Wqh, bq, nullptr, (void*)Qh, CC, CC);
  gemm_wmma<0><<<gg, 256, 0, stream>>>(kvnh, Wkh, bk, nullptr, (void*)Kh, CC, CC);
  gemm_wmma<0><<<gg, 256, 0, stream>>>(kvnh, Wvh, bv, nullptr, (void*)Vh, CC, CC);

  attn<<<BB * HH * (LL / 128), 256, 0, stream>>>(Qh, Kh, Vh, AOh);

  gemm_wmma<1><<<gg, 256, 0, stream>>>(AOh, Woh, bo, qn32, d_out, CC, CC);
}
